// MultiheadAttention_11699490914619
// MI455X (gfx1250) — compile-verified, matrix operand order checked by reading, not run on hardware
//
#include <hip/hip_runtime.h>

// ---------------------------------------------------------------------------
// MHA for B=2, S=2048, D=1024, H=16, HD=64 on gfx1250 (MI455X).
// Compute-bound (~69 GFLOP vs ~80MB HBM traffic) -> everything on the
// v_wmma_f32_16x16x32_f16 path with f32 accumulation. Intermediates (32MB of
// f16 Q/K/V heads + attention output) live in workspace; L2 (192MB) resident.
// Attention stages K/V tiles via double-buffered global_load_async_to_lds_b128
// (ASYNCcnt) and does softmax row reductions with DPP row_xmask (no LDS pipe).
// ---------------------------------------------------------------------------

typedef _Float16 v16h __attribute__((ext_vector_type(16)));
typedef _Float16 v8h  __attribute__((ext_vector_type(8)));
typedef _Float16 v4h  __attribute__((ext_vector_type(4)));
typedef float    v8f  __attribute__((ext_vector_type(8)));
typedef float    v4f  __attribute__((ext_vector_type(4)));

#define S_LEN 2048
#define D_DIM 1024
#define H_CNT 16
#define HD    64
#define M_TOT 4096   // B*S

__device__ __forceinline__ v8f wmma_f16(v16h a, v16h b, v8f c) {
  // (neg_a, A, neg_b, B, c_mod, C, reuse_a, reuse_b)
  return __builtin_amdgcn_wmma_f32_16x16x32_f16(false, a, false, b, (short)0, c,
                                                false, false);
}

// Build a v16h A/B fragment from two contiguous 8-half (16-byte) runs.
// CDNA5 16-bit fragment layout: lane holds halfs for k = base..base+7 and
// k = base+16..base+23 where base = 8*(lane>>4) within a K=32 chunk.
__device__ __forceinline__ v16h frag2(const _Float16* p0, const _Float16* p1) {
  v8h lo = *(const v8h*)p0;
  v8h hi = *(const v8h*)p1;
  return __builtin_shufflevector(lo, hi, 0, 1, 2, 3, 4, 5, 6, 7,
                                 8, 9, 10, 11, 12, 13, 14, 15);
}

// --- DPP row_xmask reductions within 16-lane rows (wave32) -----------------
template <int CTRL>
__device__ __forceinline__ float dpp_xmask(float x) {
  int xi = __builtin_bit_cast(int, x);
  int r  = __builtin_amdgcn_update_dpp(xi, xi, CTRL, 0xf, 0xf, true);
  return __builtin_bit_cast(float, r);
}
__device__ __forceinline__ float rowmax16(float x) {
  x = fmaxf(x, dpp_xmask<0x161>(x));  // ROW_XMASK:1
  x = fmaxf(x, dpp_xmask<0x162>(x));  // ROW_XMASK:2
  x = fmaxf(x, dpp_xmask<0x164>(x));  // ROW_XMASK:4
  x = fmaxf(x, dpp_xmask<0x168>(x));  // ROW_XMASK:8
  return x;
}
__device__ __forceinline__ float rowsum16(float x) {
  x += dpp_xmask<0x161>(x);
  x += dpp_xmask<0x162>(x);
  x += dpp_xmask<0x164>(x);
  x += dpp_xmask<0x168>(x);
  return x;
}

// --- async global -> LDS copy of a 64x64 f16 tile (128 threads, 4 b128 each)
// LDS row stride 72 halfs: every 16B chunk stays 16B-aligned (144 = 9*16) and
// b128 fragment reads are bank-conflict-free (36-dword stride).
#define LDP 72
__device__ __forceinline__ void async_copy_tile64(const _Float16* gbase,
                                                  int grow_stride,
                                                  unsigned lds_base, int tid) {
#pragma unroll
  for (int i = 0; i < 4; ++i) {
    int chunk = tid + i * 128;                 // 512 chunks of 8 halfs
    int row = chunk >> 3;
    int c8  = (chunk & 7) * 8;
    unsigned long long g =
        (unsigned long long)(const void*)(gbase + (size_t)row * grow_stride + c8);
    unsigned l = lds_base + (unsigned)((row * LDP + c8) * 2);
    asm volatile("global_load_async_to_lds_b128 %0, %1, off"
                 :: "v"(l), "v"(g) : "memory");
  }
}

template <typename T> struct V4T;
template <> struct V4T<float>    { using type = v4f; };
template <> struct V4T<_Float16> { using type = v4h; };

// ---------------------------------------------------------------------------
// GEMM: Out = (X[M,1024] * W[N=1024,K=1024]^T + bias) * out_scale
// Block tile 128x64, BK=32, 8 waves (wave32): wave grid 4(m) x 2(n),
// each wave owns a 2x2 array of 16x16 WMMA accumulators.
// MODE 0: f16 out, head-split   [B,H,S,64]   (Q, K)
// MODE 1: f16 out, head-split-T [B,H,64,S]   (V)
// MODE 3: f32 out, plain        [M,N]        (final projection -> d_out)
// ---------------------------------------------------------------------------
template <typename Tin, int MODE>
__global__ __launch_bounds__(256) void gemm_wmma_kernel(
    const Tin* __restrict__ X, const float* __restrict__ W,
    const float* __restrict__ bias, void* __restrict__ Out, float out_scale) {
  constexpr int K = D_DIM;
  constexpr int LDX = 40;  // 32 halfs + 8 pad (80B row stride)
  __shared__ _Float16 Xs[128 * LDX];
  __shared__ _Float16 Ws[64 * LDX];

  const int tid  = threadIdx.x;
  const int lane = tid & 31;
  const int wid  = tid >> 5;
  const int wave_m = wid & 3;   // 0..3 -> 32 rows each
  const int wave_n = wid >> 2;  // 0..1 -> 32 cols each
  const int m0 = blockIdx.y * 128;
  const int n0 = blockIdx.x * 64;
  const int lcol = lane & 15;
  const int hi8  = (lane >> 4) * 8;

  v8f acc[2][2] = {};

  for (int k0 = 0; k0 < K; k0 += 32) {
    // --- cooperative load + f16 conversion into LDS ---
    using VT = typename V4T<Tin>::type;
#pragma unroll
    for (int i = 0; i < 4; ++i) {       // X tile: 128x32
      int g   = tid + i * 256;
      int row = g >> 3;
      int c4  = (g & 7) << 2;
      VT xv = *(const VT*)(X + (size_t)(m0 + row) * K + k0 + c4);
      _Float16* dst = &Xs[row * LDX + c4];
      dst[0] = (_Float16)xv.x; dst[1] = (_Float16)xv.y;
      dst[2] = (_Float16)xv.z; dst[3] = (_Float16)xv.w;
    }
#pragma unroll
    for (int i = 0; i < 2; ++i) {       // W tile: 64x32 (always f32)
      int g   = tid + i * 256;
      int row = g >> 3;
      int c4  = (g & 7) << 2;
      v4f wv = *(const v4f*)(W + (size_t)(n0 + row) * K + k0 + c4);
      _Float16* dst = &Ws[row * LDX + c4];
      dst[0] = (_Float16)wv.x; dst[1] = (_Float16)wv.y;
      dst[2] = (_Float16)wv.z; dst[3] = (_Float16)wv.w;
    }
    // hint the next K tile toward the caches while we compute this one
    if (k0 + 32 < K) {
      __builtin_prefetch(X + (size_t)(m0 + (tid >> 3)) * K + (k0 + 32) +
                             ((tid & 7) << 2), 0, 1);
      if (tid < 128)
        __builtin_prefetch(W + (size_t)(n0 + (tid >> 1)) * K + (k0 + 32) +
                               ((tid & 1) << 4), 0, 1);
    }
    __syncthreads();

    // --- fragments + WMMA ---
    v16h af[2], bf[2];
#pragma unroll
    for (int ms = 0; ms < 2; ++ms) {
      const _Float16* p = &Xs[(wave_m * 32 + ms * 16 + lcol) * LDX];
      af[ms] = frag2(p + hi8, p + 16 + hi8);
    }
#pragma unroll
    for (int ns = 0; ns < 2; ++ns) {
      const _Float16* p = &Ws[(wave_n * 32 + ns * 16 + lcol) * LDX];
      bf[ns] = frag2(p + hi8, p + 16 + hi8);
    }
#pragma unroll
    for (int ms = 0; ms < 2; ++ms)
#pragma unroll
      for (int ns = 0; ns < 2; ++ns)
        acc[ms][ns] = wmma_f16(af[ms], bf[ns], acc[ms][ns]);
    __syncthreads();
  }

  // --- epilogue: bias, scale, scatter per layout MODE ---
#pragma unroll
  for (int ns = 0; ns < 2; ++ns) {
    const int n = n0 + wave_n * 32 + ns * 16 + lcol;
    const float bn = bias[n];
#pragma unroll
    for (int ms = 0; ms < 2; ++ms) {
#pragma unroll
      for (int r = 0; r < 8; ++r) {
        const int m = m0 + wave_m * 32 + ms * 16 + hi8 + r;
        const float val = (acc[ms][ns][r] + bn) * out_scale;
        if (MODE == 3) {
          ((float*)Out)[(size_t)m * D_DIM + n] = val;
        } else {
          const int b = m >> 11, s = m & (S_LEN - 1);
          const int h = n >> 6, d = n & (HD - 1);
          size_t idx;
          if (MODE == 0)
            idx = (((size_t)(b * H_CNT + h) * S_LEN + s) * HD + d);
          else  // MODE 1: transposed V
            idx = (((size_t)(b * H_CNT + h) * HD + d) * S_LEN + s);
          ((_Float16*)Out)[idx] = (_Float16)val;
        }
      }
    }
  }
}

// ---------------------------------------------------------------------------
// Flash attention: grid (S/64, H, B), 128 threads = 4 waves, each wave owns a
// 16-row query strip. Q pre-scaled by 1/sqrt(HD) in the projection kernel.
// K/V 64x64 tiles are double-buffered in LDS via async global->LDS b128 copies
// (next tile's copies overlap this tile's WMMAs; s_wait_asynccnt 8 retires the
// current tile since async loads complete in order). P is redistributed
// through a per-wave-private LDS strip (same-wave LDS is in-order).
// ---------------------------------------------------------------------------
__global__ __launch_bounds__(128) void attn_wmma_kernel(
    const _Float16* __restrict__ qh,  // [B,H,S,64] (scaled)
    const _Float16* __restrict__ kh,  // [B,H,S,64]
    const _Float16* __restrict__ vt,  // [B,H,64,S]
    _Float16* __restrict__ ao) {      // [B,S,H*64]
  __shared__ _Float16 Ks[2][64 * LDP];
  __shared__ _Float16 Vs[2][64 * LDP];
  __shared__ _Float16 P[64 * LDP];

  const int tid  = threadIdx.x;
  const int lane = tid & 31;
  const int wid  = tid >> 5;
  const int qb = blockIdx.x, h = blockIdx.y, b = blockIdx.z;
  const size_t head = (size_t)(b * H_CNT + h) * (S_LEN * HD);
  const int lcol = lane & 15;
  const int hi8  = (lane >> 4) * 8;

  const _Float16* kbase = kh + head;  // key rows, stride HD
  const _Float16* vbase = vt + head;  // d rows,   stride S_LEN
  const unsigned ksb[2] = {(unsigned)(size_t)&Ks[0][0],
                           (unsigned)(size_t)&Ks[1][0]};
  const unsigned vsb[2] = {(unsigned)(size_t)&Vs[0][0],
                           (unsigned)(size_t)&Vs[1][0]};

  // Q fragments for this wave's 16 rows (two K=32 chunks over head dim 64)
  const int sq = qb * 64 + wid * 16 + lcol;
  const _Float16* qrow = qh + head + (size_t)sq * HD;
  v16h qf0 = frag2(qrow + hi8,      qrow + 16 + hi8);
  v16h qf1 = frag2(qrow + 32 + hi8, qrow + 48 + hi8);

  v8f o[4] = {};
  float mrow[8], lrow[8];
#pragma unroll
  for (int r = 0; r < 8; ++r) { mrow[r] = -1e30f; lrow[r] = 0.0f; }

  constexpr int NT = S_LEN / 64;  // 32 key tiles

  // prologue: stage tile 0 into buffer 0
  async_copy_tile64(kbase, HD, ksb[0], tid);
  async_copy_tile64(vbase, S_LEN, vsb[0], tid);

  for (int kc = 0; kc < NT; ++kc) {
    const int p = kc & 1;
    if (kc + 1 < NT) {  // issue next tile into the other buffer
      const int k1 = (kc + 1) * 64;
      async_copy_tile64(kbase + (size_t)k1 * HD, HD, ksb[p ^ 1], tid);
      async_copy_tile64(vbase + k1, S_LEN, vsb[p ^ 1], tid);
      asm volatile("s_wait_asynccnt 0x8" ::: "memory");  // tile kc resident
    } else {
      asm volatile("s_wait_asynccnt 0x0" ::: "memory");
    }
    __syncthreads();  // all threads' chunks of tile kc visible

    // ---- S = Q @ K^T  (16 x 64 scores per wave) ----
    v8f sc[4] = {};
#pragma unroll
    for (int nt = 0; nt < 4; ++nt) {
      const _Float16* kr = &Ks[p][(nt * 16 + lcol) * LDP];
      v16h b0 = frag2(kr + hi8,      kr + 16 + hi8);
      v16h b1 = frag2(kr + 32 + hi8, kr + 48 + hi8);
      sc[nt] = wmma_f16(qf0, b0, sc[nt]);
      sc[nt] = wmma_f16(qf1, b1, sc[nt]);
    }

    // ---- online softmax: DPP row_xmask reductions, no LDS traffic ----
#pragma unroll
    for (int r = 0; r < 8; ++r) {
      float mx = fmaxf(fmaxf(sc[0][r], sc[1][r]), fmaxf(sc[2][r], sc[3][r]));
      mx = rowmax16(mx);
      const float mnew = fmaxf(mrow[r], mx);
      const float al   = __expf(mrow[r] - mnew);
      float rs = 0.0f;
#pragma unroll
      for (int nt = 0; nt < 4; ++nt) {
        const float pv = __expf(sc[nt][r] - mnew);
        sc[nt][r] = pv;
        rs += pv;
      }
      rs = rowsum16(rs);
      lrow[r] = lrow[r] * al + rs;
      mrow[r] = mnew;
#pragma unroll
      for (int dt = 0; dt < 4; ++dt) o[dt][r] *= al;
    }

    // ---- P -> LDS (wave-private strip; same-wave LDS is in-order) ----
#pragma unroll
    for (int nt = 0; nt < 4; ++nt)
#pragma unroll
      for (int r = 0; r < 8; ++r)
        P[(wid * 16 + hi8 + r) * LDP + nt * 16 + lcol] = (_Float16)sc[nt][r];

    const _Float16* prow = &P[(wid * 16 + lcol) * LDP];
    v16h pf0 = frag2(prow + hi8,      prow + 16 + hi8);
    v16h pf1 = frag2(prow + 32 + hi8, prow + 48 + hi8);

    // ---- O += P @ V ----
#pragma unroll
    for (int dt = 0; dt < 4; ++dt) {
      const _Float16* vr = &Vs[p][(dt * 16 + lcol) * LDP];
      v16h v0 = frag2(vr + hi8,      vr + 16 + hi8);
      v16h v1 = frag2(vr + 32 + hi8, vr + 48 + hi8);
      o[dt] = wmma_f16(pf0, v0, o[dt]);
      o[dt] = wmma_f16(pf1, v1, o[dt]);
    }
    __syncthreads();  // done reading buffer p before it is overwritten
  }

  // ---- normalize and write merged-head f16 output [B,S,D] ----
#pragma unroll
  for (int r = 0; r < 8; ++r) {
    const float inv = 1.0f / lrow[r];
    const int s = qb * 64 + wid * 16 + hi8 + r;
    const size_t rowbase = ((size_t)b * S_LEN + s) * D_DIM + h * HD;
#pragma unroll
    for (int dt = 0; dt < 4; ++dt)
      ao[rowbase + dt * 16 + lcol] = (_Float16)(o[dt][r] * inv);
  }
}

extern "C" void kernel_launch(void* const* d_in, const int* in_sizes, int n_in,
                              void* d_out, int out_size, void* d_ws,
                              size_t ws_size, hipStream_t stream) {
  (void)in_sizes; (void)n_in; (void)out_size; (void)ws_size;
  const float* q  = (const float*)d_in[0];
  const float* k  = (const float*)d_in[1];
  const float* v  = (const float*)d_in[2];
  const float* Wq = (const float*)d_in[3];
  const float* bq = (const float*)d_in[4];
  const float* Wk = (const float*)d_in[5];
  const float* bk = (const float*)d_in[6];
  const float* Wv = (const float*)d_in[7];
  const float* bv = (const float*)d_in[8];
  const float* Wo = (const float*)d_in[9];
  const float* bo = (const float*)d_in[10];

  // Workspace: 4 x (4096*1024 f16) = 32 MB
  _Float16* qh = (_Float16*)d_ws;
  _Float16* kh = qh + (size_t)M_TOT * D_DIM;
  _Float16* vt = kh + (size_t)M_TOT * D_DIM;
  _Float16* ao = vt + (size_t)M_TOT * D_DIM;

  const dim3 gemm_grid(D_DIM / 64, M_TOT / 128);  // (16, 32)
  constexpr float kScale = 0.125f;                // 1/sqrt(HD)

  gemm_wmma_kernel<float, 0><<<gemm_grid, 256, 0, stream>>>(q, Wq, bq, qh, kScale);
  gemm_wmma_kernel<float, 0><<<gemm_grid, 256, 0, stream>>>(k, Wk, bk, kh, 1.0f);
  gemm_wmma_kernel<float, 1><<<gemm_grid, 256, 0, stream>>>(v, Wv, bv, vt, 1.0f);

  attn_wmma_kernel<<<dim3(S_LEN / 64, H_CNT, 2), 128, 0, stream>>>(qh, kh, vt, ao);

  gemm_wmma_kernel<_Float16, 3><<<gemm_grid, 256, 0, stream>>>(ao, Wo, bo,
                                                               d_out, 1.0f);
}
